// MultiHeadAttention_75058848465294
// MI455X (gfx1250) — matrix operand order repaired and checked by reading, not compiled, not run on hardware
//
#include <hip/hip_runtime.h>
#include <hip/hip_bf16.h>
#include <stdint.h>

typedef __bf16 bf16;
typedef __attribute__((ext_vector_type(16))) __bf16 v16bf;
typedef __attribute__((ext_vector_type(8)))  __bf16 bf16x8;
typedef __attribute__((ext_vector_type(8)))  float  v8f;

// ---------------------------------------------------------------------------
// CDNA5 async global->LDS copy (ASYNCcnt-tracked) + waits
// ---------------------------------------------------------------------------
#define WAIT_ASYNC(n) asm volatile("s_wait_asynccnt " #n ::: "memory")

__device__ inline void async_b128(const bf16* lds_dst, const bf16* gsrc) {
  // Generic pointers to LDS carry the DS byte offset in their low 32 bits.
  const uint32_t l = (uint32_t)(uintptr_t)lds_dst;
  asm volatile("global_load_async_to_lds_b128 %0, %1, off"
               :: "v"(l), "v"((uint64_t)(uintptr_t)gsrc)
               : "memory");
}

// ---------------------------------------------------------------------------
// WMMA helpers (gfx1250, wave32): D = A(16x32 bf16) x B(32x16 bf16) + C(f32)
// ---------------------------------------------------------------------------
__device__ inline v8f wmma_bf16(v16bf a, v16bf b, v8f c) {
  return __builtin_amdgcn_wmma_f32_16x16x32_bf16(
      false, a, false, b, (short)0, c, false, false);
}

__device__ inline v16bf make_frag(bf16x8 lo, bf16x8 hi) {
  v16bf r;
#pragma unroll
  for (int i = 0; i < 8; ++i) { r[i] = lo[i]; r[i + 8] = hi[i]; }
  return r;
}

// A-matrix 16x32 fragment: lane l: row M=l&15; v0..3: K=(l>>4)*8+0..7,
// v4..7: K=16+(l>>4)*8+0..7
__device__ inline v16bf load_frag_a(const bf16* base, int lda, int lane) {
  const bf16* p = base + (lane & 15) * lda + (lane >> 4) * 8;
  return make_frag(*(const bf16x8*)p, *(const bf16x8*)(p + 16));
}

// B-matrix 32x16 fragment: lane l: col N=l&15; v0..3: K=(l>>4)*8+0..7,
// v4..7: K=16+(l>>4)*8+0..7. LDS tile stored [N][K] -> 2x16B per lane.
__device__ inline v16bf load_frag_b(const bf16* base, int ldb, int lane) {
  const bf16* p = base + (lane & 15) * ldb + (lane >> 4) * 8;
  return make_frag(*(const bf16x8*)p, *(const bf16x8*)(p + 16));
}

// One K=32 step of a 128x128 block tile: wave computes 32(M) x 64(N).
__device__ inline void gemm_compute_step(const bf16 (*As)[32], const bf16 (*Bs)[32],
                                         int wm, int wn, int lane, v8f acc[2][4]) {
  v16bf a[2], b[4];
#pragma unroll
  for (int i = 0; i < 2; ++i) a[i] = load_frag_a(&As[wm + i * 16][0], 32, lane);
#pragma unroll
  for (int j = 0; j < 4; ++j) b[j] = load_frag_b(&Bs[wn + j * 16][0], 32, lane);
#pragma unroll
  for (int i = 0; i < 2; ++i)
#pragma unroll
    for (int j = 0; j < 4; ++j) acc[i][j] = wmma_bf16(a[i], b[j], acc[i][j]);
}

// ---------------------------------------------------------------------------
// Kernel 0: fp32 -> bf16 conversion (one-shot, makes all staging byte-copies)
// ---------------------------------------------------------------------------
__global__ __launch_bounds__(256) void cvt_f32_bf16_kernel(
    const float* __restrict__ s, bf16* __restrict__ d, int n8) {
  const int i = blockIdx.x * 256 + threadIdx.x;
  if (i >= n8) return;
  const float4 a = ((const float4*)s)[i * 2 + 0];
  const float4 b = ((const float4*)s)[i * 2 + 1];
  bf16x8 o;
  o[0] = (bf16)a.x; o[1] = (bf16)a.y; o[2] = (bf16)a.z; o[3] = (bf16)a.w;
  o[4] = (bf16)b.x; o[5] = (bf16)b.y; o[6] = (bf16)b.z; o[7] = (bf16)b.w;
  ((bf16x8*)d)[i] = o;
}

// ---------------------------------------------------------------------------
// Kernel 1: fused QKV projection, async double-buffered staging.
// Y = xb @ W^T; output [B,H,S,64] bf16. grid=(4,64,3), block=256 (8 waves).
// ---------------------------------------------------------------------------
__global__ __launch_bounds__(256) void qkv_proj_kernel(
    const bf16* __restrict__ xb,
    const bf16* __restrict__ Wqb, const bf16* __restrict__ Wkb,
    const bf16* __restrict__ Wvb,
    bf16* __restrict__ Qb, bf16* __restrict__ Kb, bf16* __restrict__ Vb) {
  __shared__ __align__(16) bf16 As[2][128][32];
  __shared__ __align__(16) bf16 Ws[2][128][32];

  const bf16* W  = (blockIdx.z == 0) ? Wqb : (blockIdx.z == 1) ? Wkb : Wvb;
  bf16*       Ob = (blockIdx.z == 0) ? Qb  : (blockIdx.z == 1) ? Kb  : Vb;

  const int m0 = blockIdx.y * 128, n0 = blockIdx.x * 128;
  const int t = threadIdx.x, wid = t >> 5, lane = t & 31;
  const int wm = (wid >> 1) * 32, wn = (wid & 1) * 64;
  const int sr = t >> 2, sc = (t & 3) * 8;  // staging coords (2 passes of 64 rows)

  auto issue = [&](int buf, int k0) {
#pragma unroll
    for (int p = 0; p < 2; ++p) {
      async_b128(&As[buf][sr + p * 64][sc],
                 xb + (size_t)(m0 + sr + p * 64) * 512 + k0 + sc);
      async_b128(&Ws[buf][sr + p * 64][sc],
                 W + (size_t)(n0 + sr + p * 64) * 512 + k0 + sc);
    }
  };

  const v8f vzero = {0.f, 0.f, 0.f, 0.f, 0.f, 0.f, 0.f, 0.f};
  v8f acc[2][4];
#pragma unroll
  for (int i = 0; i < 2; ++i)
#pragma unroll
    for (int j = 0; j < 4; ++j) acc[i][j] = vzero;

  issue(0, 0);
  for (int it = 0; it < 16; ++it) {
    const int cur = it & 1;
    if (it + 1 < 16) { issue(cur ^ 1, (it + 1) * 32); WAIT_ASYNC(4); }
    else             { WAIT_ASYNC(0); }
    __syncthreads();  // all waves' current-buffer copies landed
    gemm_compute_step(As[cur], Ws[cur], wm, wn, lane, acc);
    __syncthreads();  // safe to overwrite this buffer next iteration
  }

  const int half = lane >> 4, lrow = lane & 15;
#pragma unroll
  for (int i = 0; i < 2; ++i)
#pragma unroll
    for (int j = 0; j < 4; ++j) {
      const int ng = n0 + wn + j * 16 + lrow;
      const int h = ng >> 6, d = ng & 63;
#pragma unroll
      for (int r = 0; r < 8; ++r) {
        const int mg = m0 + wm + i * 16 + 8 * half + r;
        const int b = mg >> 11, s = mg & 2047;
        Ob[(((size_t)(b * 8 + h)) * 2048 + s) * 64 + d] = (bf16)acc[i][j][r];
      }
    }
}

// ---------------------------------------------------------------------------
// Kernel 2: sigmoid attention, async double-buffered K tile + VALU V^T tile.
// O = sigmoid(Q K^T / 8 - ln 2048) @ V. grid=(16,32), block=256.
// ---------------------------------------------------------------------------
__global__ __launch_bounds__(256) void sigmoid_attn_kernel(
    const bf16* __restrict__ Qb, const bf16* __restrict__ Kb,
    const bf16* __restrict__ Vb, bf16* __restrict__ Ob) {
  constexpr int SEQ = 2048, D = 64, KC = 32;
  __shared__ __align__(16) bf16 Ks[2][KC][D];     // key-major: K[key][d]
  __shared__ __align__(16) bf16 VTs[2][D][KC];    // transposed: V^T[d][key]
  __shared__ __align__(16) bf16 Ss[8][16][KC];    // per-wave prob re-layout

  const int bh = blockIdx.y;
  const int t = threadIdx.x, wid = t >> 5, lane = t & 31;
  const int lrow = lane & 15, half = lane >> 4;
  const int qw = blockIdx.x * 128 + wid * 16;

  const bf16* Qh = Qb + (size_t)bh * SEQ * D;
  const bf16* Kh = Kb + (size_t)bh * SEQ * D;
  const bf16* Vh = Vb + (size_t)bh * SEQ * D;

  v16bf qa[2];
  {
    const bf16* p = Qh + (size_t)(qw + lrow) * D + half * 8;
    qa[0] = make_frag(*(const bf16x8*)p,        *(const bf16x8*)(p + 16));
    qa[1] = make_frag(*(const bf16x8*)(p + 32), *(const bf16x8*)(p + 48));
  }

  const int key = t >> 3, seg = (t & 7) * 8;  // staging coords
  auto stage = [&](int buf, int kt) {
    async_b128(&Ks[buf][key][seg], Kh + (size_t)(kt + key) * D + seg);
    const bf16x8 vv = *(const bf16x8*)(Vh + (size_t)(kt + key) * D + seg);
#pragma unroll
    for (int i = 0; i < 8; ++i) VTs[buf][seg + i][key] = vv[i];
  };

  const v8f vzero = {0.f, 0.f, 0.f, 0.f, 0.f, 0.f, 0.f, 0.f};
  v8f acc[4] = {vzero, vzero, vzero, vzero};
  const float bias = -7.6246189861593985f;  // -ln(2048)
  const float sc = 0.125f;                  // 1/sqrt(64)

  stage(0, 0);
  for (int it = 0; it < SEQ / KC; ++it) {
    const int cur = it & 1;
    if (it + 1 < SEQ / KC) { stage(cur ^ 1, (it + 1) * KC); WAIT_ASYNC(1); }
    else                   { WAIT_ASYNC(0); }
    __syncthreads();

    // scores: two 16x16 f32 tiles (keys 0-15, 16-31 of chunk), K-dim = d
    v8f s0 = vzero, s1 = vzero;
#pragma unroll
    for (int dc = 0; dc < 2; ++dc) {
      const v16bf b0 = load_frag_b(&Ks[cur][0][dc * 32],  D, lane);
      const v16bf b1 = load_frag_b(&Ks[cur][16][dc * 32], D, lane);
      s0 = wmma_bf16(qa[dc], b0, s0);
      s1 = wmma_bf16(qa[dc], b1, s1);
    }

    // sigmoid, then D-frag -> A-frag re-layout through per-wave LDS scratch
    bf16(*sw)[KC] = Ss[wid];
#pragma unroll
    for (int r = 0; r < 8; ++r) {
      const float p0 = 1.f / (1.f + __expf(-(s0[r] * sc + bias)));
      const float p1 = 1.f / (1.f + __expf(-(s1[r] * sc + bias)));
      sw[r + 8 * half][lrow]      = (bf16)p0;
      sw[r + 8 * half][16 + lrow] = (bf16)p1;
    }
    asm volatile("s_wait_dscnt 0" ::: "memory");  // wave-local LDS RAW
    const v16bf pa = load_frag_a(&sw[0][0], KC, lane);

    // O += attn(16x32) @ V(32x64): B[k=key][n=d] from VTs[d][key]
#pragma unroll
    for (int n = 0; n < 4; ++n) {
      const v16bf bv = load_frag_b(&VTs[cur][n * 16][0], KC, lane);
      acc[n] = wmma_bf16(pa, bv, acc[n]);
    }
    __syncthreads();
  }

  const int b = bh >> 3, h = bh & 7;
#pragma unroll
  for (int n = 0; n < 4; ++n) {
    const int d = n * 16 + lrow;
#pragma unroll
    for (int r = 0; r < 8; ++r) {
      const int s = qw + r + 8 * half;
      Ob[((size_t)(b * 2048 + s)) * 512 + h * 64 + d] = (bf16)acc[n][r];
    }
  }
}

// ---------------------------------------------------------------------------
// Kernel 3: output projection, async double-buffered. out = O @ Wo^T (fp32).
// grid=(4,64), block=256.
// ---------------------------------------------------------------------------
__global__ __launch_bounds__(256) void out_proj_kernel(
    const bf16* __restrict__ A, const bf16* __restrict__ W,
    float* __restrict__ Out) {
  __shared__ __align__(16) bf16 As[2][128][32];
  __shared__ __align__(16) bf16 Ws[2][128][32];

  const int m0 = blockIdx.y * 128, n0 = blockIdx.x * 128;
  const int t = threadIdx.x, wid = t >> 5, lane = t & 31;
  const int wm = (wid >> 1) * 32, wn = (wid & 1) * 64;
  const int sr = t >> 2, sc = (t & 3) * 8;

  auto issue = [&](int buf, int k0) {
#pragma unroll
    for (int p = 0; p < 2; ++p) {
      async_b128(&As[buf][sr + p * 64][sc],
                 A + (size_t)(m0 + sr + p * 64) * 512 + k0 + sc);
      async_b128(&Ws[buf][sr + p * 64][sc],
                 W + (size_t)(n0 + sr + p * 64) * 512 + k0 + sc);
    }
  };

  const v8f vzero = {0.f, 0.f, 0.f, 0.f, 0.f, 0.f, 0.f, 0.f};
  v8f acc[2][4];
#pragma unroll
  for (int i = 0; i < 2; ++i)
#pragma unroll
    for (int j = 0; j < 4; ++j) acc[i][j] = vzero;

  issue(0, 0);
  for (int it = 0; it < 16; ++it) {
    const int cur = it & 1;
    if (it + 1 < 16) { issue(cur ^ 1, (it + 1) * 32); WAIT_ASYNC(4); }
    else             { WAIT_ASYNC(0); }
    __syncthreads();
    gemm_compute_step(As[cur], Ws[cur], wm, wn, lane, acc);
    __syncthreads();
  }

  const int half = lane >> 4, lrow = lane & 15;
#pragma unroll
  for (int i = 0; i < 2; ++i)
#pragma unroll
    for (int j = 0; j < 4; ++j) {
      const int ng = n0 + wn + j * 16 + lrow;
#pragma unroll
      for (int r = 0; r < 8; ++r) {
        const int mg = m0 + wm + i * 16 + 8 * half + r;
        Out[(size_t)mg * 512 + ng] = acc[i][j][r];
      }
    }
}

// ---------------------------------------------------------------------------
extern "C" void kernel_launch(void* const* d_in, const int* in_sizes, int n_in,
                              void* d_out, int out_size, void* d_ws, size_t ws_size,
                              hipStream_t stream) {
  (void)in_sizes; (void)n_in; (void)out_size; (void)ws_size;
  const float* x  = (const float*)d_in[0];
  const float* Wq = (const float*)d_in[1];
  const float* Wk = (const float*)d_in[2];
  const float* Wv = (const float*)d_in[3];
  const float* Wo = (const float*)d_in[4];
  float* out = (float*)d_out;

  char* ws = (char*)d_ws;
  const size_t xN = (size_t)4 * 2048 * 512;   // 4,194,304
  const size_t wN = (size_t)512 * 512;        // 262,144
  bf16* xb  = (bf16*)ws;                       ws += xN * sizeof(bf16);
  bf16* Qb  = (bf16*)ws;                       ws += xN * sizeof(bf16);
  bf16* Kb  = (bf16*)ws;                       ws += xN * sizeof(bf16);
  bf16* Vb  = (bf16*)ws;                       ws += xN * sizeof(bf16);
  bf16* Ob  = (bf16*)ws;                       ws += xN * sizeof(bf16);
  bf16* Wqb = (bf16*)ws;                       ws += wN * sizeof(bf16);
  bf16* Wkb = (bf16*)ws;                       ws += wN * sizeof(bf16);
  bf16* Wvb = (bf16*)ws;                       ws += wN * sizeof(bf16);
  bf16* Wob = (bf16*)ws;

  cvt_f32_bf16_kernel<<<dim3((int)(xN / 8 / 256)), 256, 0, stream>>>(x, xb, (int)(xN / 8));
  cvt_f32_bf16_kernel<<<dim3((int)(wN / 8 / 256)), 256, 0, stream>>>(Wq, Wqb, (int)(wN / 8));
  cvt_f32_bf16_kernel<<<dim3((int)(wN / 8 / 256)), 256, 0, stream>>>(Wk, Wkb, (int)(wN / 8));
  cvt_f32_bf16_kernel<<<dim3((int)(wN / 8 / 256)), 256, 0, stream>>>(Wv, Wvb, (int)(wN / 8));
  cvt_f32_bf16_kernel<<<dim3((int)(wN / 8 / 256)), 256, 0, stream>>>(Wo, Wob, (int)(wN / 8));

  qkv_proj_kernel<<<dim3(4, 64, 3), 256, 0, stream>>>(xb, Wqb, Wkb, Wvb, Qb, Kb, Vb);
  sigmoid_attn_kernel<<<dim3(16, 32), 256, 0, stream>>>(Qb, Kb, Vb, Ob);
  out_proj_kernel<<<dim3(4, 64), 256, 0, stream>>>(Ob, Wob, out);
}
